// PatchSamplePoseF_41429254537850
// MI455X (gfx1250) — matrix operand order repaired and checked by reading, not compiled, not run on hardware
//
#include <hip/hip_runtime.h>
#include <hip/hip_bf16.h>

typedef __attribute__((ext_vector_type(16))) __bf16 v16bf;
typedef __attribute__((ext_vector_type(8)))  __bf16 v8bf;
typedef __attribute__((ext_vector_type(8)))  float  v8f;

#define NCOL 256      // NC
#define ROWS 32       // rows (patches) per block
#define NTHR 256      // 8 waves of 32
#define TOTROWS 32768 // B * P = 8 * 4096

union AFrag { v16bf v; v8bf h[2]; };

// Convert fp32 (C x 256) weight to bf16 transposed (256 x C), K-contiguous.
__global__ void wt_convert_transpose(const float* __restrict__ src,
                                     __bf16* __restrict__ dst, int C) {
    int idx = blockIdx.x * blockDim.x + threadIdx.x;
    int total = C * NCOL;
    if (idx >= total) return;
    int c = idx / NCOL;
    int n = idx - c * NCOL;
    dst[n * C + c] = (__bf16)src[idx];
}

template<int C, int HW>
__global__ __launch_bounds__(NTHR)
void fused_scale_kernel(const float* __restrict__ feat,   // (8, C, HW)
                        const int*   __restrict__ pid,    // (8, 4096) flat
                        const __bf16* __restrict__ W1t,   // (256, C) bf16
                        const float* __restrict__ b1,
                        const __bf16* __restrict__ W2t,   // (256, 256) bf16
                        const float* __restrict__ b2,
                        float* __restrict__ out) {        // (32768, 256)
    __shared__ __align__(32) __bf16 sH[ROWS * NCOL];   // hidden, bf16 (16KB)
    __shared__ __align__(32) float  sU[ROWS * NCOL];   // union: X(bf16) / Y(f32) (32KB)
    __shared__ float     sB1[NCOL];
    __shared__ float     sB2[NCOL];
    __shared__ float     sNorm[ROWS];
    __shared__ long long sOff[ROWS];

    const int tid  = threadIdx.x;
    const int row0 = blockIdx.x * ROWS;

    sB1[tid] = b1[tid];
    sB2[tid] = b2[tid];
    if (tid < ROWS) {
        int n = row0 + tid;           // flat patch id; b = n >> 12
        int b = n >> 12;
        int s = pid[n];               // pid flat index equals n
        sOff[tid] = (long long)b * C * HW + s;
    }
    __syncthreads();

    // ---- gather X (fp32, stride-HW channel gather) -> LDS as bf16 ----
    __bf16* sX = (__bf16*)sU;
    for (int e = tid; e < ROWS * C; e += NTHR) {
        int r = e / C;
        int c = e & (C - 1);
        float v = feat[sOff[r] + (long long)c * HW];
        sX[r * C + c] = (__bf16)v;
    }
    __syncthreads();

    const int wave = tid >> 5;
    const int lane = tid & 31;
    const int half = lane >> 4;   // which 16-lane half
    const int ln   = lane & 15;
    const int tm   = wave >> 2;        // 0..1  (ROWS/16 row-tiles)
    const int tn0  = (wave & 3) * 4;   // 4 column tiles per wave

    // ---- GEMM1: H = relu(X @ W1 + b1), K = C ----
    {
        v8f acc[4] = {};
        const int arow = (tm * 16 + ln) * C;
        #pragma unroll
        for (int k0 = 0; k0 < C; k0 += 32) {
            AFrag a;
            a.h[0] = *(const v8bf*)&sX[arow + k0 + 16 * half];       // K = k0+16h .. +7
            a.h[1] = *(const v8bf*)&sX[arow + k0 + 16 * half + 8];   // K = k0+16h+8 .. +7
            #pragma unroll
            for (int i = 0; i < 4; ++i) {
                const int tn = tn0 + i;
                v16bf bfr = *(const v16bf*)&W1t[(tn * 16 + ln) * C + k0 + 16 * half];
                acc[i] = __builtin_amdgcn_wmma_f32_16x16x32_bf16(
                    false, a.v, false, bfr, (short)0, acc[i], false, false);
            }
        }
        #pragma unroll
        for (int i = 0; i < 4; ++i) {
            const int N = (tn0 + i) * 16 + ln;
            const float bias = sB1[N];
            #pragma unroll
            for (int r = 0; r < 8; ++r) {
                const int M = tm * 16 + 8 * half + r;
                float h = fmaxf(acc[i][r] + bias, 0.f);
                sH[M * NCOL + N] = (__bf16)h;
            }
        }
    }
    __syncthreads();

    // ---- GEMM2: Y = H @ W2 + b2, K = 256; Y (f32) overwrites X region ----
    {
        v8f acc[4] = {};
        const int arow = (tm * 16 + ln) * NCOL;
        #pragma unroll
        for (int k0 = 0; k0 < NCOL; k0 += 32) {
            AFrag a;
            a.h[0] = *(const v8bf*)&sH[arow + k0 + 16 * half];
            a.h[1] = *(const v8bf*)&sH[arow + k0 + 16 * half + 8];
            #pragma unroll
            for (int i = 0; i < 4; ++i) {
                const int tn = tn0 + i;
                v16bf bfr = *(const v16bf*)&W2t[(tn * 16 + ln) * NCOL + k0 + 16 * half];
                acc[i] = __builtin_amdgcn_wmma_f32_16x16x32_bf16(
                    false, a.v, false, bfr, (short)0, acc[i], false, false);
            }
        }
        #pragma unroll
        for (int i = 0; i < 4; ++i) {
            const int N = (tn0 + i) * 16 + ln;
            const float bias = sB2[N];
            #pragma unroll
            for (int r = 0; r < 8; ++r) {
                const int M = tm * 16 + 8 * half + r;
                sU[M * NCOL + N] = acc[i][r] + bias;
            }
        }
    }
    __syncthreads();

    // ---- row L2 norms ----
    if (tid < ROWS) {
        const float* yr = &sU[tid * NCOL];
        float s = 0.f;
        #pragma unroll 8
        for (int c = 0; c < NCOL; ++c) { float v = yr[c]; s += v * v; }
        sNorm[tid] = sqrtf(s);
    }
    __syncthreads();

    // ---- normalize + coalesced float4 writeout ----
    const float4* Y4   = (const float4*)sU;
    float4*       out4 = (float4*)(out + (long long)row0 * NCOL);
    #pragma unroll
    for (int it = 0; it < (ROWS * NCOL / 4) / NTHR; ++it) {
        int i4 = tid + it * NTHR;
        int r  = i4 >> 6;                 // / (NCOL/4)
        float inv = 1.f / (sNorm[r] + 1e-7f);
        float4 v = Y4[i4];
        v.x *= inv; v.y *= inv; v.z *= inv; v.w *= inv;
        out4[i4] = v;
    }
}

extern "C" void kernel_launch(void* const* d_in, const int* in_sizes, int n_in,
                              void* d_out, int out_size, void* d_ws, size_t ws_size,
                              hipStream_t stream) {
    // input order: feat0,pid0,w1_0,b1_0,w2_0,b2_0, feat1,..., feat2,..., num_patches
    const float* feat0 = (const float*)d_in[0];
    const int*   pid0  = (const int*)  d_in[1];
    const float* w1_0  = (const float*)d_in[2];
    const float* b1_0  = (const float*)d_in[3];
    const float* w2_0  = (const float*)d_in[4];
    const float* b2_0  = (const float*)d_in[5];
    const float* feat1 = (const float*)d_in[6];
    const int*   pid1  = (const int*)  d_in[7];
    const float* w1_1  = (const float*)d_in[8];
    const float* b1_1  = (const float*)d_in[9];
    const float* w2_1  = (const float*)d_in[10];
    const float* b2_1  = (const float*)d_in[11];
    const float* feat2 = (const float*)d_in[12];
    const int*   pid2  = (const int*)  d_in[13];
    const float* w1_2  = (const float*)d_in[14];
    const float* b1_2  = (const float*)d_in[15];
    const float* w2_2  = (const float*)d_in[16];
    const float* b2_2  = (const float*)d_in[17];

    float* out = (float*)d_out;

    // bf16 transposed-weight workspace layout (element offsets)
    __bf16* ws = (__bf16*)d_ws;
    __bf16* W1t0 = ws;                    // 256*128
    __bf16* W2t0 = ws + 32768;            // 256*256
    __bf16* W1t1 = ws + 98304;            // 256*256
    __bf16* W2t1 = ws + 163840;           // 256*256
    __bf16* W1t2 = ws + 229376;           // 256*512
    __bf16* W2t2 = ws + 360448;           // 256*256

    // weight conversion (tiny)
    wt_convert_transpose<<<(128 * 256 + 255) / 256, 256, 0, stream>>>(w1_0, W1t0, 128);
    wt_convert_transpose<<<(256 * 256 + 255) / 256, 256, 0, stream>>>(w2_0, W2t0, 256);
    wt_convert_transpose<<<(256 * 256 + 255) / 256, 256, 0, stream>>>(w1_1, W1t1, 256);
    wt_convert_transpose<<<(256 * 256 + 255) / 256, 256, 0, stream>>>(w2_1, W2t1, 256);
    wt_convert_transpose<<<(512 * 256 + 255) / 256, 256, 0, stream>>>(w1_2, W1t2, 512);
    wt_convert_transpose<<<(256 * 256 + 255) / 256, 256, 0, stream>>>(w2_2, W2t2, 256);

    const int nblk = TOTROWS / ROWS;  // 1024
    fused_scale_kernel<128, 256 * 128><<<nblk, NTHR, 0, stream>>>(
        feat0, pid0, W1t0, b1_0, W2t0, b2_0, out);
    fused_scale_kernel<256, 128 * 64><<<nblk, NTHR, 0, stream>>>(
        feat1, pid1, W1t1, b1_1, W2t1, b2_1, out + (long long)TOTROWS * NCOL);
    fused_scale_kernel<512, 64 * 32><<<nblk, NTHR, 0, stream>>>(
        feat2, pid2, W1t2, b1_2, W2t2, b2_2, out + 2LL * TOTROWS * NCOL);
}
